// BiDAF_22797686407639
// MI455X (gfx1250) — matrix operand order repaired and checked by reading, not compiled, not run on hardware
//
#include <hip/hip_runtime.h>
#include <hip/hip_bf16.h>

// ---------------- types ----------------
typedef __bf16 bf16;
typedef __attribute__((ext_vector_type(16))) __bf16 v16bf;
typedef __attribute__((ext_vector_type(8)))  float  v8f;

// ---------------- model constants ----------------
#define BB    64
#define CLEN  400
#define QLEN  48
#define TCH   16
#define CDIM  16
#define CCH   100
#define CW    5
#define WDIM  300
#define HDdim 400
#define HH    400          // lstm hidden
#define NC    (BB*CLEN)    // 25600
#define NQ    (BB*QLEN)    // 3072
#define KP400 416          // 400 padded to mult of 32

static __device__ __forceinline__ float sigmf(float x){ return 1.0f/(1.0f+expf(-x)); }

// ---------------- WMMA fragment loads ----------------
// A (16x32 bf16): lane = {half = lane>>4, r = lane&15}; row M = r,
// e[0..7]  = K kb+half*8 .. +7 ; e[8..15] = K kb+16+half*8 .. +7
static __device__ __forceinline__ v16bf load_a_frag(const bf16* __restrict__ A, int lda,
                                                    int row, int kb, int half) {
  union { v16bf v; uint4 q[2]; } f;
  const bf16* p = A + (size_t)row * lda + kb + half * 8;
  f.q[0] = *(const uint4*)(p);
  f.q[1] = *(const uint4*)(p + 16);
  return f.v;
}
// B (32x16 bf16) from W[N,K] row-major: lane column n = r, e[0..7] = K kb+half*8 .. +7 ; e[8..15] = K kb+16+half*8 .. +7
static __device__ __forceinline__ v16bf load_b_frag(const bf16* __restrict__ W, int ldw,
                                                    int nrow, int kb, int half) {
  union { v16bf v; uint4 q[2]; } f;
  const bf16* p = W + (size_t)nrow * ldw + kb + half * 8;
  f.q[0] = *(const uint4*)(p);
  f.q[1] = *(const uint4*)(p + 16);
  return f.v;
}

// ---------------- generic batched GEMM: C = act(A @ W^T + bias) ----------------
// A: [M,K] bf16 (lda=K, mult of 32), W: [N,K] bf16, C: [M,N] f32.
// M mult of 16, N mult of 16. Each wave: 1 M-tile x up to 4 N-tiles.
// ACT: 0 none, 1 sigmoid, 2 relu
template <int ACT>
__global__ __launch_bounds__(256) void k_gemm(
    const bf16* __restrict__ A, long long sA, int lda,
    const bf16* __restrict__ W, long long sW, int ldw,
    const float* __restrict__ bias,
    float* __restrict__ C, long long sC, int ldc,
    int M, int N, int K)
{
  int bat = blockIdx.y;
  A += (size_t)bat * sA; W += (size_t)bat * sW; C += (size_t)bat * sC;
  int wave = threadIdx.x >> 5, lane = threadIdx.x & 31;
  int half = lane >> 4, r = lane & 15;
  int Mt = M >> 4, Ng = (N + 63) >> 6;
  int group = blockIdx.x * 8 + wave;
  if (group >= Mt * Ng) return;
  int mt = group % Mt, ng = group / Mt;
  int m0 = mt << 4, n0 = ng << 6;
  int nt = (N - n0) >> 4; if (nt > 4) nt = 4;

  v8f acc[4] = {v8f{0}, v8f{0}, v8f{0}, v8f{0}};
  for (int kb = 0; kb < K; kb += 32) {
    v16bf a = load_a_frag(A, lda, m0 + r, kb, half);
    #pragma unroll
    for (int j = 0; j < 4; ++j) {
      if (j < nt) {
        v16bf b = load_b_frag(W, ldw, n0 + 16 * j + r, kb, half);
        acc[j] = __builtin_amdgcn_wmma_f32_16x16x32_bf16(false, a, false, b,
                                                         (short)0, acc[j], false, false);
      }
    }
  }
  for (int j = 0; j < nt; ++j) {
    int n = n0 + 16 * j + r;
    float bb = bias ? bias[n] : 0.0f;
    #pragma unroll
    for (int v = 0; v < 8; ++v) {
      float x = acc[j][v] + bb;
      if (ACT == 1) x = sigmf(x);
      if (ACT == 2) x = fmaxf(x, 0.0f);
      C[(size_t)(m0 + v + half * 8) * ldc + n] = x;
    }
  }
}

// ---------------- fused fwd+bwd LSTM step ----------------
// xw*: [B*T,1600] f32 precomputed x@Wih^T+b.  Whh*: [1600, 416] bf16 (K padded).
// hcur/hnew: [2][64][416] bf16 (pad zeros).  cst: [2][64][400] f32.
// hs: [B*T, 800] f32 output (fwd cols 0:400, bwd 400:800).
__global__ __launch_bounds__(256) void k_lstm_step(
    const float* __restrict__ xwf, const float* __restrict__ xwb,
    const bf16* __restrict__ Whhf, const bf16* __restrict__ Whhb,
    const bf16* __restrict__ hcur, bf16* __restrict__ hnew,
    float* __restrict__ cst, float* __restrict__ hs, int T, int t)
{
  int wave = threadIdx.x >> 5, lane = threadIdx.x & 31;
  int half = lane >> 4, r = lane & 15;
  int wf = blockIdx.x * 8 + wave;            // 0..199 exactly (grid=25 blocks)
  int dir = wf / 100; int rest = wf % 100;
  int mt = rest & 3, jt = rest >> 2;         // mt: batch tile 0..3, jt: hidden tile 0..24
  const bf16*  Whh = dir ? Whhb : Whhf;
  const float* xw  = dir ? xwb  : xwf;
  const bf16*  h   = hcur + (size_t)dir * 64 * KP400;

  v8f acc[4] = {v8f{0}, v8f{0}, v8f{0}, v8f{0}};
  for (int kb = 0; kb < KP400; kb += 32) {
    v16bf a = load_a_frag(h, KP400, mt * 16 + r, kb, half);
    #pragma unroll
    for (int g = 0; g < 4; ++g) {
      v16bf b = load_b_frag(Whh, KP400, g * HH + jt * 16 + r, kb, half);
      acc[g] = __builtin_amdgcn_wmma_f32_16x16x32_bf16(false, a, false, b,
                                                       (short)0, acc[g], false, false);
    }
  }
  int tt = dir ? (T - 1 - t) : t;
  int j = jt * 16 + r;
  #pragma unroll
  for (int v = 0; v < 8; ++v) {
    int b = mt * 16 + v + half * 8;
    size_t row = (size_t)b * T + tt;
    float gi = acc[0][v] + xw[row * 1600 + j];
    float gf = acc[1][v] + xw[row * 1600 + 400 + j];
    float gg = acc[2][v] + xw[row * 1600 + 800 + j];
    float go = acc[3][v] + xw[row * 1600 + 1200 + j];
    float* cp = cst + ((size_t)dir * 64 + b) * HH + j;
    float cn = sigmf(gf) * (*cp) + sigmf(gi) * tanhf(gg);
    float hn = sigmf(go) * tanhf(cn);
    *cp = cn;
    hs[row * 800 + dir * HH + j] = hn;
    hnew[((size_t)dir * 64 + b) * KP400 + j] = (bf16)hn;
  }
}

// ---------------- f32 -> bf16 with K padding (also used for weights [N,K]) ----------------
__global__ void k_cvt(const float* __restrict__ src, bf16* __restrict__ dst,
                      int N, int K, int Kp)
{
  long long idx = (long long)blockIdx.x * blockDim.x + threadIdx.x;
  if (idx >= (long long)N * Kp) return;
  int n = (int)(idx / Kp), k = (int)(idx % Kp);
  dst[idx] = (k < K) ? (bf16)src[(size_t)n * K + k] : (bf16)0.0f;
}

// ---------------- char CNN: embed + conv(width5) + maxpool ----------------
__global__ __launch_bounds__(256) void k_charcnn(
    const int* __restrict__ ids, const float* __restrict__ emb,
    const float* __restrict__ Wc, const float* __restrict__ bc,
    float* __restrict__ out, int Npos)
{
  __shared__ float sW[CCH * CW * CDIM];   // 100*80 floats = 32KB
  __shared__ float sE[8][TCH * CDIM];     // 8 positions * 256 floats
  for (int i = threadIdx.x; i < CCH * CW * CDIM; i += 256) sW[i] = Wc[i];
  int wave = threadIdx.x >> 5, lane = threadIdx.x & 31;
  int p = blockIdx.x * 8 + wave;
  if (p < Npos) {
    for (int i = lane; i < TCH * CDIM; i += 32) {
      int ch = i >> 4, d = i & 15;
      int id = ids[(size_t)p * TCH + ch];
      sE[wave][i] = emb[(size_t)id * CDIM + d];
    }
  }
  __syncthreads();
  if (p < Npos) {
    for (int f = lane; f < CCH; f += 32) {
      float best = -1e30f;
      float bias = bc[f];
      for (int t = 0; t < TCH - CW + 1; ++t) {
        float s = bias;
        const float* w = &sW[f * CW * CDIM];
        const float* e = &sE[wave][t * CDIM];
        #pragma unroll
        for (int x = 0; x < CW * CDIM; ++x) s += w[x] * e[x];
        best = fmaxf(best, s);
      }
      out[(size_t)p * CCH + f] = best;
    }
  }
}

// ---------------- x = [char_feat | glove[word]] ----------------
__global__ void k_build_x(const float* __restrict__ cc, const int* __restrict__ word,
                          const float* __restrict__ glove, float* __restrict__ x, long long n)
{
  long long idx = (long long)blockIdx.x * blockDim.x + threadIdx.x;
  if (idx >= n) return;
  int p = (int)(idx / HDdim), d = (int)(idx % HDdim);
  x[idx] = (d < CCH) ? cc[(size_t)p * CCH + d]
                     : glove[(size_t)word[p] * WDIM + (d - CCH)];
}

// ---------------- highway combine: x = g*nl + (1-g)*lin ; also write padded bf16 ----------------
__global__ void k_hw_combine(const float* __restrict__ tg, const float* __restrict__ tn,
                             const float* __restrict__ tl, float* __restrict__ x,
                             bf16* __restrict__ xbf, int Np)
{
  long long idx = (long long)blockIdx.x * blockDim.x + threadIdx.x;
  if (idx >= (long long)Np * KP400) return;
  int p = (int)(idx / KP400), k = (int)(idx % KP400);
  if (k < HDdim) {
    size_t o = (size_t)p * HDdim + k;
    float g = tg[o], v = g * tn[o] + (1.0f - g) * tl[o];
    x[o] = v;
    xbf[idx] = (bf16)v;
  } else {
    xbf[idx] = (bf16)0.0f;
  }
}

// ---------------- attention preps ----------------
__global__ __launch_bounds__(256) void k_att_c(const float* __restrict__ c,
                                               const float* __restrict__ attw,
                                               float* __restrict__ cwc,
                                               bf16* __restrict__ cm, int Np)
{
  int wave = threadIdx.x >> 5, lane = threadIdx.x & 31;
  int p = blockIdx.x * 8 + wave; if (p >= Np) return;
  float s = 0.0f;
  for (int d = lane; d < 800; d += 32) {
    float v = c[(size_t)p * 800 + d];
    s += v * attw[d];                               // wc
    cm[(size_t)p * 800 + d] = (bf16)(v * attw[1600 + d]);  // wm
  }
  for (int o = 16; o > 0; o >>= 1) s += __shfl_xor(s, o, 32);
  if (lane == 0) cwc[p] = s;
}

__global__ __launch_bounds__(256) void k_att_q(const float* __restrict__ q,
                                               const float* __restrict__ attw,
                                               float* __restrict__ qwq, int Np)
{
  int wave = threadIdx.x >> 5, lane = threadIdx.x & 31;
  int p = blockIdx.x * 8 + wave; if (p >= Np) return;
  float s = 0.0f;
  for (int d = lane; d < 800; d += 32) s += q[(size_t)p * 800 + d] * attw[800 + d];
  for (int o = 16; o > 0; o >>= 1) s += __shfl_xor(s, o, 32);
  if (lane == 0) qwq[p] = s;
}

// softmax over 48 q positions; one wave per (b,i) row; S1 stored bf16 [p,64] (pad zeros)
__global__ __launch_bounds__(256) void k_softmax48(
    const float* __restrict__ Sdot, const float* __restrict__ cwc,
    const float* __restrict__ qwq, const float* __restrict__ attb,
    bf16* __restrict__ S1, float* __restrict__ rowmax, int Np)
{
  int wave = threadIdx.x >> 5, lane = threadIdx.x & 31;
  int p = blockIdx.x * 8 + wave; if (p >= Np) return;
  int b = p / CLEN;
  float base = cwc[p] + attb[0];
  int j0 = lane, j1 = lane + 32;
  float v0 = Sdot[(size_t)p * 48 + j0] + qwq[(size_t)b * 48 + j0] + base;
  float v1 = (j1 < 48) ? (Sdot[(size_t)p * 48 + j1] + qwq[(size_t)b * 48 + j1] + base) : -1e30f;
  float m = fmaxf(v0, v1);
  for (int o = 16; o > 0; o >>= 1) m = fmaxf(m, __shfl_xor(m, o, 32));
  float e0 = expf(v0 - m), e1 = (j1 < 48) ? expf(v1 - m) : 0.0f;
  float s = e0 + e1;
  for (int o = 16; o > 0; o >>= 1) s += __shfl_xor(s, o, 32);
  float inv = 1.0f / s;
  S1[(size_t)p * 64 + j0] = (bf16)(e0 * inv);
  S1[(size_t)p * 64 + j1] = (j1 < 48) ? (bf16)(e1 * inv) : (bf16)0.0f;
  if (lane == 0) rowmax[p] = m;
}

// qT[b][n][j] = q[b][j][n], j padded 48->64
__global__ void k_qT(const float* __restrict__ q, bf16* __restrict__ qT)
{
  long long idx = (long long)blockIdx.x * blockDim.x + threadIdx.x;
  if (idx >= (long long)BB * 800 * 64) return;
  int b = (int)(idx / (800 * 64));
  int r = (int)(idx % (800 * 64));
  int n = r / 64, j = r % 64;
  qT[idx] = (j < 48) ? (bf16)q[((size_t)b * 48 + j) * 800 + n] : (bf16)0.0f;
}

// S2 softmax over context + weighted sum h[b,d] = sum_i w_i c[b,i,d]
__global__ __launch_bounds__(256) void k_s2h(const float* __restrict__ rowmax,
                                             const float* __restrict__ c,
                                             float* __restrict__ h)
{
  int b = blockIdx.x, tid = threadIdx.x;
  __shared__ float red[256];
  __shared__ float wv[CLEN];
  float m = -1e30f;
  for (int i = tid; i < CLEN; i += 256) m = fmaxf(m, rowmax[(size_t)b * CLEN + i]);
  red[tid] = m; __syncthreads();
  for (int o = 128; o > 0; o >>= 1) { if (tid < o) red[tid] = fmaxf(red[tid], red[tid + o]); __syncthreads(); }
  m = red[0]; __syncthreads();
  float s = 0.0f;
  for (int i = tid; i < CLEN; i += 256) {
    float e = expf(rowmax[(size_t)b * CLEN + i] - m);
    wv[i] = e; s += e;
  }
  red[tid] = s; __syncthreads();
  for (int o = 128; o > 0; o >>= 1) { if (tid < o) red[tid] += red[tid + o]; __syncthreads(); }
  float inv = 1.0f / red[0]; __syncthreads();
  for (int d = tid; d < 800; d += 256) {
    float a = 0.0f;
    for (int i = 0; i < CLEN; ++i) a += wv[i] * c[((size_t)b * CLEN + i) * 800 + d];
    h[(size_t)b * 800 + d] = a * inv;
  }
}

// g = [c, U, c*U, c*c*h]  (f32 + bf16)
__global__ void k_build_g(const float* __restrict__ c, const float* __restrict__ U,
                          const float* __restrict__ hv, float* __restrict__ g,
                          bf16* __restrict__ gbf)
{
  long long idx = (long long)blockIdx.x * blockDim.x + threadIdx.x;
  if (idx >= (long long)NC * 3200) return;
  int p = (int)(idx / 3200), d = (int)(idx % 3200);
  int b = p / CLEN;
  float v;
  if (d < 800)        v = c[(size_t)p * 800 + d];
  else if (d < 1600)  v = U[(size_t)p * 800 + (d - 800)];
  else if (d < 2400) { int dd = d - 1600; v = c[(size_t)p * 800 + dd] * U[(size_t)p * 800 + dd]; }
  else               { int dd = d - 2400; float cv = c[(size_t)p * 800 + dd];
                       v = cv * cv * hv[(size_t)b * 800 + dd]; }
  g[idx] = v;
  gbf[idx] = (bf16)v;
}

// logits[p] = dot(g[p],w[0:3200]) + dot(m[p],w[3200:4000]) + b
__global__ __launch_bounds__(256) void k_logits(const float* __restrict__ g,
                                                const float* __restrict__ m,
                                                const float* __restrict__ w,
                                                const float* __restrict__ bptr,
                                                float* __restrict__ out, int Np)
{
  int wave = threadIdx.x >> 5, lane = threadIdx.x & 31;
  int p = blockIdx.x * 8 + wave; if (p >= Np) return;
  float s = 0.0f;
  for (int d = lane; d < 3200; d += 32) s += g[(size_t)p * 3200 + d] * w[d];
  for (int d = lane; d < 800;  d += 32) s += m[(size_t)p * 800 + d] * w[3200 + d];
  for (int o = 16; o > 0; o >>= 1) s += __shfl_xor(s, o, 32);
  if (lane == 0) out[p] = s + bptr[0];
}

// log_softmax over CLEN per batch
__global__ __launch_bounds__(256) void k_logsoftmax(const float* __restrict__ lg,
                                                    float* __restrict__ out)
{
  int b = blockIdx.x, tid = threadIdx.x;
  __shared__ float red[256];
  float m = -1e30f;
  for (int i = tid; i < CLEN; i += 256) m = fmaxf(m, lg[(size_t)b * CLEN + i]);
  red[tid] = m; __syncthreads();
  for (int o = 128; o > 0; o >>= 1) { if (tid < o) red[tid] = fmaxf(red[tid], red[tid + o]); __syncthreads(); }
  m = red[0]; __syncthreads();
  float s = 0.0f;
  for (int i = tid; i < CLEN; i += 256) s += expf(lg[(size_t)b * CLEN + i] - m);
  red[tid] = s; __syncthreads();
  for (int o = 128; o > 0; o >>= 1) { if (tid < o) red[tid] += red[tid + o]; __syncthreads(); }
  float ls = logf(red[0]);
  for (int i = tid; i < CLEN; i += 256)
    out[(size_t)b * CLEN + i] = lg[(size_t)b * CLEN + i] - m - ls;
}

// ================= host orchestration =================
static inline long long cdiv(long long a, long long b) { return (a + b - 1) / b; }

extern "C" void kernel_launch(void* const* d_in, const int* in_sizes, int n_in,
                              void* d_out, int out_size, void* d_ws, size_t ws_size,
                              hipStream_t stream)
{
  (void)in_sizes; (void)n_in; (void)out_size; (void)ws_size;
  // ---- inputs (setup_inputs dict order, params flattened depth-first) ----
  const int*   c_char = (const int*)d_in[0];
  const int*   q_char = (const int*)d_in[1];
  const int*   c_word = (const int*)d_in[2];
  const int*   q_word = (const int*)d_in[3];
  const float* char_emb = (const float*)d_in[4];
  const float* conv_W   = (const float*)d_in[5];
  const float* conv_b   = (const float*)d_in[6];
  const float* glove    = (const float*)d_in[7];
  const float* hw_p[12]; for (int i = 0; i < 12; ++i) hw_p[i] = (const float*)d_in[8 + i];
  const float* att_w = (const float*)d_in[20];
  const float* att_b = (const float*)d_in[21];
  const float* ctx_p[6];  for (int i = 0; i < 6; ++i) ctx_p[i]  = (const float*)d_in[22 + i];
  const float* mod0_p[6]; for (int i = 0; i < 6; ++i) mod0_p[i] = (const float*)d_in[28 + i];
  const float* mod1_p[6]; for (int i = 0; i < 6; ++i) mod1_p[i] = (const float*)d_in[34 + i];
  const float* out_p[6];  for (int i = 0; i < 6; ++i) out_p[i]  = (const float*)d_in[40 + i];
  const float* p1_w = (const float*)d_in[46];
  const float* p1_b = (const float*)d_in[47];
  const float* p2_w = (const float*)d_in[48];
  const float* p2_b = (const float*)d_in[49];
  float* outp = (float*)d_out;

  // ---- workspace bump allocator ----
  char* ws = (char*)d_ws; size_t off = 0;
  auto alloc = [&](size_t bytes) -> void* {
    void* p = ws + off; off = (off + bytes + 255) & ~(size_t)255; return p;
  };
  float* cc     = (float*)alloc((size_t)NC * CCH * 4);
  float* qc     = (float*)alloc((size_t)NQ * CCH * 4);
  float* xc     = (float*)alloc((size_t)NC * HDdim * 4);
  float* xq     = (float*)alloc((size_t)NQ * HDdim * 4);
  bf16*  xc_bf  = (bf16*) alloc((size_t)NC * KP400 * 2);
  bf16*  xq_bf  = (bf16*) alloc((size_t)NQ * KP400 * 2);
  float* tg     = (float*)alloc((size_t)NC * HDdim * 4);
  float* tn     = (float*)alloc((size_t)NC * HDdim * 4);
  float* tl     = (float*)alloc((size_t)NC * HDdim * 4);
  float* c_enc  = (float*)alloc((size_t)NC * 800 * 4);
  float* q_enc  = (float*)alloc((size_t)NQ * 800 * 4);
  bf16*  c_encb = (bf16*) alloc((size_t)NC * 800 * 2);
  bf16*  q_encb = (bf16*) alloc((size_t)NQ * 800 * 2);
  float* xwf    = (float*)alloc((size_t)NC * 1600 * 4);
  float* xwb    = (float*)alloc((size_t)NC * 1600 * 4);
  float* cstate = (float*)alloc((size_t)2 * 64 * HH * 4);
  bf16*  hbufA  = (bf16*) alloc((size_t)2 * 64 * KP400 * 2);
  bf16*  hbufB  = (bf16*) alloc((size_t)2 * 64 * KP400 * 2);
  float* m0     = (float*)alloc((size_t)NC * 800 * 4);
  bf16*  m0_bf  = (bf16*) alloc((size_t)NC * 800 * 2);
  float* mm     = (float*)alloc((size_t)NC * 800 * 4);
  bf16*  mm_bf  = (bf16*) alloc((size_t)NC * 800 * 2);
  float* m2     = (float*)alloc((size_t)NC * 800 * 4);
  float* Ubuf   = (float*)alloc((size_t)NC * 800 * 4);
  float* gbuf   = (float*)alloc((size_t)NC * 3200 * 4);
  bf16*  g_bf   = (bf16*) alloc((size_t)NC * 3200 * 2);
  float* Sdot   = (float*)alloc((size_t)NC * 48 * 4);
  bf16*  S1_bf  = (bf16*) alloc((size_t)NC * 64 * 2);
  bf16*  qT_bf  = (bf16*) alloc((size_t)BB * 800 * 64 * 2);
  bf16*  cm_bf  = (bf16*) alloc((size_t)NC * 800 * 2);
  float* cwc    = (float*)alloc((size_t)NC * 4);
  float* qwq    = (float*)alloc((size_t)NQ * 4);
  float* rowmax = (float*)alloc((size_t)NC * 4);
  float* hvec   = (float*)alloc((size_t)BB * 800 * 4);
  float* logit  = (float*)alloc((size_t)NC * 4);
  // bf16 weights
  bf16* whw[6];
  for (int i = 0; i < 6; ++i) whw[i] = (bf16*)alloc((size_t)HDdim * KP400 * 2);
  bf16* wctx_ihf = (bf16*)alloc((size_t)1600 * KP400 * 2);
  bf16* wctx_hhf = (bf16*)alloc((size_t)1600 * KP400 * 2);
  bf16* wctx_ihb = (bf16*)alloc((size_t)1600 * KP400 * 2);
  bf16* wctx_hhb = (bf16*)alloc((size_t)1600 * KP400 * 2);
  bf16* wm0_ihf  = (bf16*)alloc((size_t)1600 * 3200 * 2);
  bf16* wm0_hhf  = (bf16*)alloc((size_t)1600 * KP400 * 2);
  bf16* wm0_ihb  = (bf16*)alloc((size_t)1600 * 3200 * 2);
  bf16* wm0_hhb  = (bf16*)alloc((size_t)1600 * KP400 * 2);
  bf16* wm1_ihf  = (bf16*)alloc((size_t)1600 * 800 * 2);
  bf16* wm1_hhf  = (bf16*)alloc((size_t)1600 * KP400 * 2);
  bf16* wm1_ihb  = (bf16*)alloc((size_t)1600 * 800 * 2);
  bf16* wm1_hhb  = (bf16*)alloc((size_t)1600 * KP400 * 2);
  bf16* wo_ihf   = (bf16*)alloc((size_t)1600 * 800 * 2);
  bf16* wo_hhf   = (bf16*)alloc((size_t)1600 * KP400 * 2);
  bf16* wo_ihb   = (bf16*)alloc((size_t)1600 * 800 * 2);
  bf16* wo_hhb   = (bf16*)alloc((size_t)1600 * KP400 * 2);

  // ---- helpers ----
  auto cvt = [&](const float* src, bf16* dst, int N, int K, int Kp) {
    long long n = (long long)N * Kp;
    k_cvt<<<(unsigned)cdiv(n, 256), 256, 0, stream>>>(src, dst, N, K, Kp);
  };
  auto gemm = [&](int act, const bf16* A, long long sA, int lda,
                  const bf16* W, long long sW, int ldw, const float* bias,
                  float* C, long long sC, int ldc, int M, int N, int K, int batch) {
    int Mt = M / 16, Ng = (N + 63) / 64;
    dim3 grid((unsigned)cdiv((long long)Mt * Ng, 8), batch);
    if (act == 0)      k_gemm<0><<<grid, 256, 0, stream>>>(A, sA, lda, W, sW, ldw, bias, C, sC, ldc, M, N, K);
    else if (act == 1) k_gemm<1><<<grid, 256, 0, stream>>>(A, sA, lda, W, sW, ldw, bias, C, sC, ldc, M, N, K);
    else               k_gemm<2><<<grid, 256, 0, stream>>>(A, sA, lda, W, sW, ldw, bias, C, sC, ldc, M, N, K);
  };
  auto lstm = [&](const float* xf, const float* xb, const bf16* whf, const bf16* whb,
                  float* hs, int T) {
    hipMemsetAsync(hbufA, 0, (size_t)2 * 64 * KP400 * 2, stream);
    hipMemsetAsync(hbufB, 0, (size_t)2 * 64 * KP400 * 2, stream);
    hipMemsetAsync(cstate, 0, (size_t)2 * 64 * HH * 4, stream);
    bf16* hc = hbufA; bf16* hn = hbufB;
    for (int t = 0; t < T; ++t) {
      k_lstm_step<<<25, 256, 0, stream>>>(xf, xb, whf, whb, hc, hn, cstate, hs, T, t);
      bf16* tmp = hc; hc = hn; hn = tmp;
    }
  };

  // ---- weight conversion ----
  for (int l = 0; l < 2; ++l)
    for (int t = 0; t < 3; ++t)
      cvt(hw_p[l * 6 + t * 2], whw[l * 3 + t], HDdim, HDdim, KP400);
  cvt(ctx_p[0], wctx_ihf, 1600, 400, KP400);  cvt(ctx_p[1], wctx_hhf, 1600, 400, KP400);
  cvt(ctx_p[3], wctx_ihb, 1600, 400, KP400);  cvt(ctx_p[4], wctx_hhb, 1600, 400, KP400);
  cvt(mod0_p[0], wm0_ihf, 1600, 3200, 3200);  cvt(mod0_p[1], wm0_hhf, 1600, 400, KP400);
  cvt(mod0_p[3], wm0_ihb, 1600, 3200, 3200);  cvt(mod0_p[4], wm0_hhb, 1600, 400, KP400);
  cvt(mod1_p[0], wm1_ihf, 1600, 800, 800);    cvt(mod1_p[1], wm1_hhf, 1600, 400, KP400);
  cvt(mod1_p[3], wm1_ihb, 1600, 800, 800);    cvt(mod1_p[4], wm1_hhb, 1600, 400, KP400);
  cvt(out_p[0],  wo_ihf,  1600, 800, 800);    cvt(out_p[1],  wo_hhf,  1600, 400, KP400);
  cvt(out_p[3],  wo_ihb,  1600, 800, 800);    cvt(out_p[4],  wo_hhb,  1600, 400, KP400);

  // ---- char CNN + embedding concat ----
  k_charcnn<<<(unsigned)cdiv(NC, 8), 256, 0, stream>>>(c_char, char_emb, conv_W, conv_b, cc, NC);
  k_charcnn<<<(unsigned)cdiv(NQ, 8), 256, 0, stream>>>(q_char, char_emb, conv_W, conv_b, qc, NQ);
  k_build_x<<<(unsigned)cdiv((long long)NC * HDdim, 256), 256, 0, stream>>>(cc, c_word, glove, xc, (long long)NC * HDdim);
  k_build_x<<<(unsigned)cdiv((long long)NQ * HDdim, 256), 256, 0, stream>>>(qc, q_word, glove, xq, (long long)NQ * HDdim);
  cvt(xc, xc_bf, NC, HDdim, KP400);
  cvt(xq, xq_bf, NQ, HDdim, KP400);

  // ---- highway (shared weights for c and q) ----
  auto highway = [&](float* x, bf16* xbf, int Np) {
    for (int l = 0; l < 2; ++l) {
      gemm(1, xbf, 0, KP400, whw[l * 3 + 0], 0, KP400, hw_p[l * 6 + 1], tg, 0, HDdim, Np, HDdim, KP400, 1);
      gemm(2, xbf, 0, KP400, whw[l * 3 + 1], 0, KP400, hw_p[l * 6 + 3], tn, 0, HDdim, Np, HDdim, KP400, 1);
      gemm(0, xbf, 0, KP400, whw[l * 3 + 2], 0, KP400, hw_p[l * 6 + 5], tl, 0, HDdim, Np, HDdim, KP400, 1);
      k_hw_combine<<<(unsigned)cdiv((long long)Np * KP400, 256), 256, 0, stream>>>(tg, tn, tl, x, xbf, Np);
    }
  };
  highway(xc, xc_bf, NC);
  highway(xq, xq_bf, NQ);

  // ---- ctx BiLSTM ----
  gemm(0, xc_bf, 0, KP400, wctx_ihf, 0, KP400, ctx_p[2], xwf, 0, 1600, NC, 1600, KP400, 1);
  gemm(0, xc_bf, 0, KP400, wctx_ihb, 0, KP400, ctx_p[5], xwb, 0, 1600, NC, 1600, KP400, 1);
  lstm(xwf, xwb, wctx_hhf, wctx_hhb, c_enc, CLEN);
  gemm(0, xq_bf, 0, KP400, wctx_ihf, 0, KP400, ctx_p[2], xwf, 0, 1600, NQ, 1600, KP400, 1);
  gemm(0, xq_bf, 0, KP400, wctx_ihb, 0, KP400, ctx_p[5], xwb, 0, 1600, NQ, 1600, KP400, 1);
  lstm(xwf, xwb, wctx_hhf, wctx_hhb, q_enc, QLEN);
  cvt(c_enc, c_encb, NC, 800, 800);
  cvt(q_enc, q_encb, NQ, 800, 800);

  // ---- attention ----
  k_att_c<<<(unsigned)cdiv(NC, 8), 256, 0, stream>>>(c_enc, att_w, cwc, cm_bf, NC);
  k_att_q<<<(unsigned)cdiv(NQ, 8), 256, 0, stream>>>(q_enc, att_w, qwq, NQ);
  gemm(0, cm_bf, (long long)CLEN * 800, 800, q_encb, (long long)QLEN * 800, 800, nullptr,
       Sdot, (long long)CLEN * 48, 48, CLEN, 48, 800, BB);
  k_softmax48<<<(unsigned)cdiv(NC, 8), 256, 0, stream>>>(Sdot, cwc, qwq, att_b, S1_bf, rowmax, NC);
  k_qT<<<(unsigned)cdiv((long long)BB * 800 * 64, 256), 256, 0, stream>>>(q_enc, qT_bf);
  gemm(0, S1_bf, (long long)CLEN * 64, 64, qT_bf, (long long)800 * 64, 64, nullptr,
       Ubuf, (long long)CLEN * 800, 800, CLEN, 800, 64, BB);
  k_s2h<<<BB, 256, 0, stream>>>(rowmax, c_enc, hvec);
  k_build_g<<<(unsigned)cdiv((long long)NC * 3200, 256), 256, 0, stream>>>(c_enc, Ubuf, hvec, gbuf, g_bf);

  // ---- modeling BiLSTMs ----
  gemm(0, g_bf, 0, 3200, wm0_ihf, 0, 3200, mod0_p[2], xwf, 0, 1600, NC, 1600, 3200, 1);
  gemm(0, g_bf, 0, 3200, wm0_ihb, 0, 3200, mod0_p[5], xwb, 0, 1600, NC, 1600, 3200, 1);
  lstm(xwf, xwb, wm0_hhf, wm0_hhb, m0, CLEN);
  cvt(m0, m0_bf, NC, 800, 800);
  gemm(0, m0_bf, 0, 800, wm1_ihf, 0, 800, mod1_p[2], xwf, 0, 1600, NC, 1600, 800, 1);
  gemm(0, m0_bf, 0, 800, wm1_ihb, 0, 800, mod1_p[5], xwb, 0, 1600, NC, 1600, 800, 1);
  lstm(xwf, xwb, wm1_hhf, wm1_hhb, mm, CLEN);
  cvt(mm, mm_bf, NC, 800, 800);

  // ---- p1 ----
  k_logits<<<(unsigned)cdiv(NC, 8), 256, 0, stream>>>(gbuf, mm, p1_w, p1_b, logit, NC);
  k_logsoftmax<<<BB, 256, 0, stream>>>(logit, outp);

  // ---- output BiLSTM + p2 ----
  gemm(0, mm_bf, 0, 800, wo_ihf, 0, 800, out_p[2], xwf, 0, 1600, NC, 1600, 800, 1);
  gemm(0, mm_bf, 0, 800, wo_ihb, 0, 800, out_p[5], xwb, 0, 1600, NC, 1600, 800, 1);
  lstm(xwf, xwb, wo_hhf, wo_hhb, m2, CLEN);
  k_logits<<<(unsigned)cdiv(NC, 8), 256, 0, stream>>>(gbuf, m2, p2_w, p2_b, logit, NC);
  k_logsoftmax<<<BB, 256, 0, stream>>>(logit, outp + NC);
}
